// BandedMultiheadedAttention_63230508532535
// MI455X (gfx1250) — matrix operand order repaired and checked by reading, not compiled, not run on hardware
//
#include <hip/hip_runtime.h>
#include <hip/hip_bf16.h>

// ---------------- problem constants ----------------
#define D_MODEL 1024
#define D_INT   64
#define KW      32
#define BATCH   2
#define SEQ     4096
#define SUBH    5
#define HEADS   14
#define MROWS   (BATCH * SEQ)          // 8192
#define NQ      (SUBH * D_INT)         // 320
#define NV      (HEADS * D_INT)        // 896

// ---------------- WMMA vector types ----------------
typedef __attribute__((ext_vector_type(16))) __bf16 v16bf;
typedef __attribute__((ext_vector_type(8)))  float  v8f;
typedef __attribute__((ext_vector_type(4)))  __bf16 bf16x4;

union V16 { v16bf v; uint4 q[2]; };   // fragment <-> 2 x 16B LDS loads

// =====================================================================
// Phase 1 & 3: tiled GEMM  C[M,N] = A[M,K] * W + bias(broadcast over M)
// Weights addressed in segments: col n -> seg = n/segN, elem (k, n%segN)
// at W[seg*K*segN + k*segN + (n%segN)].
// Tile: 128(M) x 64(N), K-step 32, 256 threads = 8 wave32.
// Pipeline per K-step: barrier -> issue ALL fragment ds_load_b128 into
// distinct registers -> issue global fetch of tile k+1 (covers LDS
// latency) -> 4 back-to-back WMMAs. Ping-pong LDS, ONE barrier/step.
// A in LDS: row-major, stride 40 (80B: 16B-aligned, 20-bank stride).
// B in LDS: fragment-ordered [t][lane][j] (2 x ds_load_b128 / frag).
// =====================================================================
#define BM 128
#define BN 64
#define BK 32
#define ASTR 40
#define ASZ (BM * ASTR)        // bf16 elements per A buffer
#define BSZ (4 * 32 * 16)      // bf16 elements per B buffer

__global__ __launch_bounds__(256) void gemm_bf16_wmma(
    const float* __restrict__ A, const float* __restrict__ W,
    const float* __restrict__ bias, float* __restrict__ C,
    int M, int N, int K, int segN)
{
    __shared__ __align__(16) __bf16 sA[2 * ASZ];      // 2 x 10240 B
    __shared__ __align__(16) __bf16 sB[2 * BSZ];      // 2 x  4096 B

    const int tid      = threadIdx.x;
    const int wave     = tid >> 5;
    const int lane     = tid & 31;
    const int lane15   = lane & 15;
    const int laneHalf = lane >> 4;

    const int nTiles = N / BN;
    const int m0 = ((int)blockIdx.x / nTiles) * BM;
    const int n0 = ((int)blockIdx.x % nTiles) * BN;

    // weight segment for this N-tile (BN=64 never straddles a segment)
    const int seg  = n0 / segN;
    const int nin0 = n0 % segN;
    const float* __restrict__ Wseg = W + (size_t)seg * (size_t)K * (size_t)segN;

    // accumulators pre-loaded with bias (C element N = lane15 + t*16)
    v8f acc[4];
#pragma unroll
    for (int t = 0; t < 4; ++t) {
        const float bvv = bias[n0 + t * 16 + lane15];
#pragma unroll
        for (int r = 0; r < 8; ++r) acc[t][r] = bvv;
    }

    // staging thread mappings
    const int ac4 = tid & 7;                  // A: 8 float4 cols per row
    const int ar0 = tid >> 3;                 // A: 32 rows per pass
    const int bn  = tid & 63;                 // B: column n within tile
    const int bkg = (tid >> 6) * 4;           // B: first k of this thread's group
    const int bt  = bn >> 4;                  // B: N-subtile
    const int bl15 = bn & 15;

    // hoisted, strided base pointers (advance by BK / BK*segN per step)
    const float* __restrict__ aPtr = A + (size_t)(m0 + ar0) * K + ac4 * 4;
    const float* __restrict__ bPtr = Wseg + (size_t)bkg * segN + nin0 + bn;
    const long aRowStride = (long)32 * K;     // 32 rows between passes
    const long bKStride   = (long)segN;

    // LDS store addresses (A: b64-aligned; B: fragment slot)
    __bf16* const sAst = &sA[ar0 * ASTR + ac4 * 4];
    const int blrLo = ((bkg & 31) >= 16 ? 16 : 0) + bl15;
    const int blrHi = (((bkg + 16) & 31) >= 16 ? 16 : 0) + bl15;
    __bf16* const sBst0 = &sB[(bt * 32 + blrLo) * 16 + (bkg & 15)];
    __bf16* const sBst1 = &sB[(bt * 32 + blrHi) * 16 + ((bkg + 16) & 15)];

    // WMMA fragment read addresses
    const __bf16* const sAfr = &sA[(wave * 16 + lane15) * ASTR + laneHalf * 16];
    const __bf16* const sBfr = &sB[lane * 16];

    // ---------------- software pipeline ----------------
    float4 aReg[4];
    float  bReg[2][4];

    // fetch tile 0
#pragma unroll
    for (int p = 0; p < 4; ++p)
        aReg[p] = *(const float4*)(aPtr + p * aRowStride);
#pragma unroll
    for (int p = 0; p < 2; ++p) {
        const float* wp = bPtr + (size_t)(p * 16) * segN;
        bReg[p][0] = wp[0];
        bReg[p][1] = wp[bKStride];
        bReg[p][2] = wp[2 * bKStride];
        bReg[p][3] = wp[3 * bKStride];
    }
    aPtr += BK;
    bPtr += (size_t)BK * segN;

    int buf = 0;
#pragma unroll 2
    for (int k0 = 0; k0 < K; k0 += BK) {
        const int aOff = buf * ASZ;
        const int bOff = buf * BSZ;

        // ---- convert + store staged registers into LDS[buf]
#pragma unroll
        for (int p = 0; p < 4; ++p) {
            bf16x4 pk;
            pk[0] = (__bf16)aReg[p].x; pk[1] = (__bf16)aReg[p].y;
            pk[2] = (__bf16)aReg[p].z; pk[3] = (__bf16)aReg[p].w;
            *(bf16x4*)(sAst + aOff + p * 32 * ASTR) = pk;
        }
#pragma unroll
        for (int p = 0; p < 2; ++p) {
            bf16x4 pk;
            pk[0] = (__bf16)bReg[p][0]; pk[1] = (__bf16)bReg[p][1];
            pk[2] = (__bf16)bReg[p][2]; pk[3] = (__bf16)bReg[p][3];
            *(bf16x4*)((p ? sBst1 : sBst0) + bOff) = pk;
        }
        __syncthreads();

        // ---- issue ALL fragment loads (distinct registers -> no per-WMMA
        //      full dscnt drain; loads overlap the global fetch below)
        V16 fa, fb0, fb1, fb2, fb3;
        fa.q[0]  = *(const uint4*)(sAfr + aOff);        // K = kb+16h..kb+16h+7
        fa.q[1]  = *(const uint4*)(sAfr + aOff + 8);    // K = kb+16h+8..kb+16h+15
        fb0.q[0] = *(const uint4*)(sBfr + bOff);
        fb0.q[1] = *(const uint4*)(sBfr + bOff + 8);
        fb1.q[0] = *(const uint4*)(sBfr + bOff + 1 * (32 * 16));
        fb1.q[1] = *(const uint4*)(sBfr + bOff + 1 * (32 * 16) + 8);
        fb2.q[0] = *(const uint4*)(sBfr + bOff + 2 * (32 * 16));
        fb2.q[1] = *(const uint4*)(sBfr + bOff + 2 * (32 * 16) + 8);
        fb3.q[0] = *(const uint4*)(sBfr + bOff + 3 * (32 * 16));
        fb3.q[1] = *(const uint4*)(sBfr + bOff + 3 * (32 * 16) + 8);

        // ---- fetch tile k+1 into registers (independent of LDS results)
        if (k0 + BK < K) {
#pragma unroll
            for (int p = 0; p < 4; ++p)
                aReg[p] = *(const float4*)(aPtr + p * aRowStride);
#pragma unroll
            for (int p = 0; p < 2; ++p) {
                const float* wp = bPtr + (size_t)(p * 16) * segN;
                bReg[p][0] = wp[0];
                bReg[p][1] = wp[bKStride];
                bReg[p][2] = wp[2 * bKStride];
                bReg[p][3] = wp[3 * bKStride];
            }
            aPtr += BK;
            bPtr += (size_t)BK * segN;
            if (k0 + 2 * BK < K)                       // warm L2 two tiles out
                __builtin_prefetch(aPtr + BK, 0, 1);
        }

        // ---- 4 back-to-back WMMAs across the 64-wide N tile
        acc[0] = __builtin_amdgcn_wmma_f32_16x16x32_bf16(
            false, fa.v, false, fb0.v, (short)0, acc[0], false, false);
        acc[1] = __builtin_amdgcn_wmma_f32_16x16x32_bf16(
            false, fa.v, false, fb1.v, (short)0, acc[1], false, false);
        acc[2] = __builtin_amdgcn_wmma_f32_16x16x32_bf16(
            false, fa.v, false, fb2.v, (short)0, acc[2], false, false);
        acc[3] = __builtin_amdgcn_wmma_f32_16x16x32_bf16(
            false, fa.v, false, fb3.v, (short)0, acc[3], false, false);
        buf ^= 1;
    }

    // ---- store: C layout => M = r + 8*laneHalf, N = lane15 (+16t)
    const int mOut = m0 + wave * 16 + laneHalf * 8;
#pragma unroll
    for (int t = 0; t < 4; ++t) {
        const int n = n0 + t * 16 + lane15;
#pragma unroll
        for (int r = 0; r < 8; ++r)
            C[(size_t)(mOut + r) * N + n] = acc[t][r];
    }
}

// =====================================================================
// Phase 2: dilated banded attention. One wave32 per (head, batch, row l).
// =====================================================================
__global__ __launch_bounds__(256) void banded_attn(
    const float* __restrict__ Qp,  // [MROWS, 320]  col = s*64+i
    const float* __restrict__ Kp,  // [MROWS, 320]
    const float* __restrict__ Vp,  // [MROWS, 896]  col = h*64+i
    const float* __restrict__ Ws,  // [14,32,32]
    const float* __restrict__ bs,  // [14,32]
    float* __restrict__ OH)        // [MROWS, 896]
{
    const int lane  = threadIdx.x & 31;
    const int gwave = (int)blockIdx.x * 8 + (threadIdx.x >> 5);   // 0 .. H*B*L-1

    const int l  = gwave % SEQ;
    const int hb = gwave / SEQ;
    const int b  = hb % BATCH;
    const int h  = hb / BATCH;

    // HEAD_OF_SUB = [0]*5 + [1]*5 + [2]*2 + [3] + [4]; DILATIONS = 1,1,2,4,8
    const int s   = (h < 10) ? (h / 5) : ((h < 12) ? 2 : (h - 9));
    const int dil = (s < 2) ? 1 : (1 << (s - 1));

    // dilated, edge-clamped window index for this lane's slot
    int idx = l + (lane - KW / 2) * dil;
    idx = idx < 0 ? 0 : (idx > SEQ - 1 ? SEQ - 1 : idx);

    const size_t rowQ = (size_t)(b * SEQ + l)   * NQ + s * D_INT;
    const size_t rowK = (size_t)(b * SEQ + idx) * NQ + s * D_INT;

    // banded score: q[l] . k[idx] / sqrt(64)
    float sc = 0.f;
#pragma unroll
    for (int i = 0; i < D_INT; i += 4) {
        const float4 qv = *(const float4*)(Qp + rowQ + i);
        const float4 kv = *(const float4*)(Kp + rowK + i);
        sc += qv.x * kv.x + qv.y * kv.y + qv.z * kv.z + qv.w * kv.w;
    }
    sc *= 0.125f;

    // positional supersampling: sampled[m=lane] = sum_k score[k] * Ws[h,k,m] + bs
    const float* __restrict__ wsh = Ws + h * KW * KW;
    float sm = bs[h * KW + lane];
#pragma unroll 8
    for (int k = 0; k < KW; ++k)
        sm += __shfl(sc, k, 32) * wsh[k * KW + lane];

    // softmax across the 32 lanes
    float mx = sm;
#pragma unroll
    for (int o = 16; o > 0; o >>= 1) mx = fmaxf(mx, __shfl_xor(mx, o, 32));
    const float e = __expf(sm - mx);
    float den = e;
#pragma unroll
    for (int o = 16; o > 0; o >>= 1) den += __shfl_xor(den, o, 32);
    const float attn = e / den;

    // out[i] = sum_k attn[k] * V[idx_k][h*64+i]; lane covers i = lane, lane+32
    float o0 = 0.f, o1 = 0.f;
#pragma unroll 8
    for (int k = 0; k < KW; ++k) {
        const float ak = __shfl(attn, k, 32);
        const int   ik = __shfl(idx,  k, 32);
        const float* __restrict__ vrow = Vp + (size_t)(b * SEQ + ik) * NV + h * D_INT;
        o0 += ak * vrow[lane];
        o1 += ak * vrow[lane + 32];
    }
    float* __restrict__ orow = OH + (size_t)(b * SEQ + l) * NV + h * D_INT;
    orow[lane]      = o0;
    orow[lane + 32] = o1;
}

// =====================================================================
extern "C" void kernel_launch(void* const* d_in, const int* in_sizes, int n_in,
                              void* d_out, int out_size, void* d_ws, size_t ws_size,
                              hipStream_t stream) {
    const float* query = (const float*)d_in[0];
    const float* key   = (const float*)d_in[1];
    const float* value = (const float*)d_in[2];
    const float* Wq    = (const float*)d_in[3];
    const float* bq    = (const float*)d_in[4];
    const float* Wk    = (const float*)d_in[5];
    const float* bk    = (const float*)d_in[6];
    const float* Wv    = (const float*)d_in[7];
    const float* bv    = (const float*)d_in[8];
    const float* Ws    = (const float*)d_in[9];
    const float* bs    = (const float*)d_in[10];
    const float* Wc    = (const float*)d_in[11];
    const float* bc    = (const float*)d_in[12];
    float* out = (float*)d_out;

    // workspace layout (floats): Qp[8192,320] Kp[8192,320] Vp[8192,896] OH[8192,896]
    float* Qp = (float*)d_ws;
    float* Kp = Qp + (size_t)MROWS * NQ;
    float* Vp = Kp + (size_t)MROWS * NQ;
    float* OH = Vp + (size_t)MROWS * NV;

    const dim3 blk(256);

    // Phase 1: projections (bf16 WMMA GEMMs, fp32 accumulate)
    gemm_bf16_wmma<<<dim3((MROWS / BM) * (NQ / BN)), blk, 0, stream>>>(
        query, Wq, bq, Qp, MROWS, NQ, D_MODEL, D_INT);
    gemm_bf16_wmma<<<dim3((MROWS / BM) * (NQ / BN)), blk, 0, stream>>>(
        key, Wk, bk, Kp, MROWS, NQ, D_MODEL, D_INT);
    gemm_bf16_wmma<<<dim3((MROWS / BM) * (NV / BN)), blk, 0, stream>>>(
        value, Wv, bv, Vp, MROWS, NV, D_MODEL, D_INT);

    // Phase 2: banded attention (one wave per head x batch x row)
    const int nWaves = HEADS * BATCH * SEQ;             // 114688
    banded_attn<<<dim3(nWaves / 8), blk, 0, stream>>>(Qp, Kp, Vp, Ws, bs, OH);

    // Phase 3: output projection [8192,896] x [896,1024] + bc
    gemm_bf16_wmma<<<dim3((MROWS / BM) * (D_MODEL / BN)), blk, 0, stream>>>(
        OH, Wc, bc, out, MROWS, D_MODEL, NV, D_MODEL);
}
